// Model_25563645346483
// MI455X (gfx1250) — compile-verified, matrix operand order checked by reading, not run on hardware
//
#include <hip/hip_runtime.h>
#include <cstdint>
#include <cstddef>

typedef __attribute__((ext_vector_type(16))) __bf16 v16bf;
typedef __attribute__((ext_vector_type(8)))  float  v8f;

#define HOPS 30

__device__ __forceinline__ float4 fma4(float4 v, float s, float4 a) {
  float4 r;
  r.x = fmaf(v.x, s, a.x); r.y = fmaf(v.y, s, a.y);
  r.z = fmaf(v.z, s, a.z); r.w = fmaf(v.w, s, a.w);
  return r;
}
__device__ __forceinline__ float4 add4(float4 a, float4 b) {
  float4 r; r.x = a.x + b.x; r.y = a.y + b.y; r.z = a.z + b.z; r.w = a.w + b.w;
  return r;
}

// ---------------- CSR construction ----------------

__global__ void k_init_deg(int* __restrict__ deg, int n) {
  int i = blockIdx.x * blockDim.x + threadIdx.x;
  if (i < n) deg[i] = 1;                      // self-loop
}

__global__ void k_count(const int* __restrict__ dst, int* __restrict__ deg, int e) {
  int i = blockIdx.x * blockDim.x + threadIdx.x;
  if (i < e) atomicAdd(&deg[dst[i]], 1);
}

// single-block exclusive scan (Hillis-Steele over 1024-chunks with running carry)
__global__ void k_scan(const int* __restrict__ deg, int* __restrict__ row_ptr,
                       int* __restrict__ cursor, int n) {
  __shared__ int buf[1024];
  __shared__ int carry_s;
  if (threadIdx.x == 0) carry_s = 0;
  __syncthreads();
  for (int base = 0; base < n; base += 1024) {
    int i = base + (int)threadIdx.x;
    int v = (i < n) ? deg[i] : 0;
    buf[threadIdx.x] = v;
    __syncthreads();
    for (int off = 1; off < 1024; off <<= 1) {
      int t = (threadIdx.x >= (unsigned)off) ? buf[threadIdx.x - off] : 0;
      __syncthreads();
      buf[threadIdx.x] += t;
      __syncthreads();
    }
    int carry = carry_s;
    int excl  = carry + buf[threadIdx.x] - v;
    if (i < n) { row_ptr[i] = excl; cursor[i] = excl; }
    __syncthreads();
    if (threadIdx.x == 1023) carry_s = carry + buf[1023];
    __syncthreads();
  }
  if (threadIdx.x == 0) row_ptr[n] = carry_s;
}

__global__ void k_dinv(const int* __restrict__ deg, float* __restrict__ dinv, int n) {
  int i = blockIdx.x * blockDim.x + threadIdx.x;
  if (i < n) dinv[i] = rsqrtf((float)deg[i]);   // deg >= 1 always (self-loop)
}

// scatter real edges (e < E) and self-loops (e >= E) into CSR slots
__global__ void k_scatter(const int* __restrict__ src, const int* __restrict__ dst,
                          const float* __restrict__ dinv, int* __restrict__ cursor,
                          int* __restrict__ csr_src, float* __restrict__ csr_nrm,
                          int E, int n) {
  int e = blockIdx.x * blockDim.x + threadIdx.x;
  if (e >= E + n) return;
  int s, d;
  if (e < E) { s = src[e]; d = dst[e]; } else { s = d = e - E; }
  int pos = atomicAdd(&cursor[d], 1);
  csr_src[pos] = s;
  csr_nrm[pos] = dinv[s] * dinv[d];
}

// ---------------- WMMA GEMM: C = act(A @ B + bias) ----------------
// A: MxK f32 row-major, B: KxNd f32 row-major, C: MxNd f32 row-major.
// Each wave computes one 16-row block x NT adjacent 16-col tiles (A fragment
// reused across NT WMMAs per K-step). 4 waves per block. K%32==0, Nd%(16*NT)==0.

template<int NT, bool RIN, bool ROUT>
__global__ void k_gemm_wmma(const float* __restrict__ A, const float* __restrict__ B,
                            const float* __restrict__ bias, float* __restrict__ C,
                            int M, int K, int Nd) {
  int lane = (int)(threadIdx.x & 31);
  int wave = (int)(threadIdx.x >> 5);
  int ngrp = Nd / (16 * NT);
  int tiles_m = (M + 15) >> 4;
  int tile = blockIdx.x * 4 + wave;          // wave-uniform
  if (tile >= tiles_m * ngrp) return;        // wave-uniform exit: EXEC stays full
  int tm = tile / ngrp, tg = tile % ngrp;

  int half = lane >> 4, idx = lane & 15;
  int arow = tm * 16 + idx; if (arow >= M) arow = M - 1;   // clamp; extras not stored
  const float* ap = A + (size_t)arow * K;
  int col0 = tg * (16 * NT) + idx;

  v8f acc[NT] = {};
  for (int kb = 0; kb < K; kb += 32) {
    int k0 = kb + half * 8;                  // multiple of 8 -> 32B-aligned float4s
    const float4* ap4 = (const float4*)(ap + k0);
    float4 q0 = ap4[0];                      // k0 .. k0+3
    float4 q1 = ap4[1];                      // k0+4 .. k0+7
    float4 q2 = ap4[4];                      // k0+16 .. k0+19
    float4 q3 = ap4[5];                      // k0+20 .. k0+23
    if (RIN) {
      q0.x = fmaxf(q0.x, 0.f); q0.y = fmaxf(q0.y, 0.f); q0.z = fmaxf(q0.z, 0.f); q0.w = fmaxf(q0.w, 0.f);
      q1.x = fmaxf(q1.x, 0.f); q1.y = fmaxf(q1.y, 0.f); q1.z = fmaxf(q1.z, 0.f); q1.w = fmaxf(q1.w, 0.f);
      q2.x = fmaxf(q2.x, 0.f); q2.y = fmaxf(q2.y, 0.f); q2.z = fmaxf(q2.z, 0.f); q2.w = fmaxf(q2.w, 0.f);
      q3.x = fmaxf(q3.x, 0.f); q3.y = fmaxf(q3.y, 0.f); q3.z = fmaxf(q3.z, 0.f); q3.w = fmaxf(q3.w, 0.f);
    }
    v16bf a;
    a[0]  = (__bf16)q0.x; a[1]  = (__bf16)q0.y; a[2]  = (__bf16)q0.z; a[3]  = (__bf16)q0.w;
    a[4]  = (__bf16)q1.x; a[5]  = (__bf16)q1.y; a[6]  = (__bf16)q1.z; a[7]  = (__bf16)q1.w;
    a[8]  = (__bf16)q2.x; a[9]  = (__bf16)q2.y; a[10] = (__bf16)q2.z; a[11] = (__bf16)q2.w;
    a[12] = (__bf16)q3.x; a[13] = (__bf16)q3.y; a[14] = (__bf16)q3.z; a[15] = (__bf16)q3.w;

#pragma unroll
    for (int t = 0; t < NT; ++t) {
      int col = col0 + t * 16;
      v16bf b;
#pragma unroll
      for (int j = 0; j < 8; ++j) {
        b[j]     = (__bf16)B[(size_t)(k0 + j)      * Nd + col];
        b[j + 8] = (__bf16)B[(size_t)(k0 + 16 + j) * Nd + col];
      }
      acc[t] = __builtin_amdgcn_wmma_f32_16x16x32_bf16(false, a, false, b,
                                                       (short)0, acc[t], false, false);
    }
  }

  bool full = (tm * 16 + 16) <= M;           // wave-uniform -> scalar branch
#pragma unroll
  for (int t = 0; t < NT; ++t) {
    int col = col0 + t * 16;
    float bv = bias[col];
    if (full) {
      float* cp = C + (size_t)(tm * 16 + 8 * half) * Nd + col;
#pragma unroll
      for (int r = 0; r < 8; ++r) {
        float v = acc[t][r] + bv;
        if (ROUT) v = fmaxf(v, 0.f);
        cp[(size_t)r * Nd] = v;
      }
    } else {
#pragma unroll
      for (int r = 0; r < 8; ++r) {
        int m = tm * 16 + r + 8 * half;
        if (m < M) {
          float v = acc[t][r] + bv;
          if (ROUT) v = fmaxf(v, 0.f);
          C[(size_t)m * Nd + col] = v;
        }
      }
    }
  }
}

// ---------------- CSR gather hop (no atomics): hout[d,:] = sum_e nrm*hin[src,:] ----
// One wave per node; each lane owns a float4 of features (Fq = F/4 = blockDim = 32
// for HID=128). Edge metadata staged through LDS in 32-entry chunks; 4 independent
// accumulator chains. Single-wave workgroup -> barriers are wave-local (S_NOP cost).

__global__ void k_hop4(const float4* __restrict__ hin, float4* __restrict__ hout,
                       const int* __restrict__ row_ptr, const int* __restrict__ csr_src,
                       const float* __restrict__ csr_nrm, int Fq) {
  __shared__ int   s_src[32];
  __shared__ float s_nrm[32];
  int node = blockIdx.x;
  int lane = (int)threadIdx.x;               // 0..Fq-1 (Fq == 32)
  int beg = row_ptr[node], end = row_ptr[node + 1];
  float4 a0 = {0,0,0,0}, a1 = {0,0,0,0}, a2 = {0,0,0,0}, a3 = {0,0,0,0};
  for (int p0 = beg; p0 < end; p0 += 32) {
    int cnt = end - p0; if (cnt > 32) cnt = 32;
    if (lane < cnt) {
      s_src[lane] = csr_src[p0 + lane];
      s_nrm[lane] = csr_nrm[p0 + lane];
    }
    __syncthreads();
    int j = 0;
    for (; j + 4 <= cnt; j += 4) {
      float4 v0 = hin[(size_t)s_src[j]     * Fq + lane];
      float4 v1 = hin[(size_t)s_src[j + 1] * Fq + lane];
      float4 v2 = hin[(size_t)s_src[j + 2] * Fq + lane];
      float4 v3 = hin[(size_t)s_src[j + 3] * Fq + lane];
      a0 = fma4(v0, s_nrm[j],     a0);
      a1 = fma4(v1, s_nrm[j + 1], a1);
      a2 = fma4(v2, s_nrm[j + 2], a2);
      a3 = fma4(v3, s_nrm[j + 3], a3);
    }
    for (; j < cnt; ++j)
      a0 = fma4(hin[(size_t)s_src[j] * Fq + lane], s_nrm[j], a0);
    __syncthreads();
  }
  hout[(size_t)node * Fq + lane] = add4(add4(a0, a1), add4(a2, a3));
}

// narrow-feature hop (OUT=16 -> Fq=4): 32 threads, 32/Fq nodes per wave
__global__ void k_hop_small4(const float4* __restrict__ hin, float4* __restrict__ hout,
                             const int* __restrict__ row_ptr, const int* __restrict__ csr_src,
                             const float* __restrict__ csr_nrm, int n, int Fq) {
  int per  = 32 / Fq;
  int node = blockIdx.x * per + (int)(threadIdx.x / Fq);
  int q    = (int)(threadIdx.x % Fq);
  if (node >= n) return;
  float4 a0 = {0,0,0,0}, a1 = {0,0,0,0};
  int beg = row_ptr[node], end = row_ptr[node + 1];
  int p = beg;
  for (; p + 2 <= end; p += 2) {
    float4 v0 = hin[(size_t)csr_src[p]     * Fq + q];
    float4 v1 = hin[(size_t)csr_src[p + 1] * Fq + q];
    a0 = fma4(v0, csr_nrm[p],     a0);
    a1 = fma4(v1, csr_nrm[p + 1], a1);
  }
  if (p < end)
    a0 = fma4(hin[(size_t)csr_src[p] * Fq + q], csr_nrm[p], a0);
  hout[(size_t)node * Fq + q] = add4(a0, a1);
}

// ---------------- driver ----------------

extern "C" void kernel_launch(void* const* d_in, const int* in_sizes, int n_in,
                              void* d_out, int out_size, void* d_ws, size_t ws_size,
                              hipStream_t stream) {
  const float* x  = (const float*)d_in[0];
  const float* W1 = (const float*)d_in[1];
  const float* b1 = (const float*)d_in[2];
  const float* W2 = (const float*)d_in[3];
  const float* b2 = (const float*)d_in[4];
  const float* Wc = (const float*)d_in[5];
  const float* bc = (const float*)d_in[6];
  const int*   ei = (const int*)d_in[7];
  // d_in[8] = conv_time (device scalar, statically 30; hardcoded for graph capture)

  const int HID = in_sizes[2];            // 128
  const int OUT = in_sizes[6];            // 16
  const int IN  = in_sizes[1] / HID;      // 256
  const int Nn  = in_sizes[0] / IN;       // 50000
  const int E   = in_sizes[7] / 2;        // 800000
  const int* srcp = ei;
  const int* dstp = ei + E;

  char* w = (char*)d_ws;
  auto bump = [&](size_t bytes) -> char* {
    char* p = w; w += (bytes + 255) & ~(size_t)255; return p;
  };
  int*   deg     = (int*)  bump(sizeof(int)   * Nn);
  int*   cursor  = (int*)  bump(sizeof(int)   * Nn);
  int*   row_ptr = (int*)  bump(sizeof(int)   * (Nn + 1));
  float* dinv    = (float*)bump(sizeof(float) * Nn);
  int*   csr_src = (int*)  bump(sizeof(int)   * (size_t)(E + Nn));
  float* csr_nrm = (float*)bump(sizeof(float) * (size_t)(E + Nn));
  float* hA      = (float*)bump(sizeof(float) * (size_t)Nn * HID);
  float* hB      = (float*)bump(sizeof(float) * (size_t)Nn * HID);
  float* h3      = (float*)bump(sizeof(float) * (size_t)Nn * OUT);

  // GCN norm + CSR (one-time)
  k_init_deg<<<(Nn + 255) / 256, 256, 0, stream>>>(deg, Nn);
  k_count  <<<(E + 255) / 256, 256, 0, stream>>>(dstp, deg, E);
  k_scan   <<<1, 1024, 0, stream>>>(deg, row_ptr, cursor, Nn);
  k_dinv   <<<(Nn + 255) / 256, 256, 0, stream>>>(deg, dinv, Nn);
  k_scatter<<<(E + Nn + 255) / 256, 256, 0, stream>>>(srcp, dstp, dinv, cursor,
                                                      csr_src, csr_nrm, E, Nn);

  int tiles_m = (Nn + 15) / 16;

  // conv1: hA = relu(x @ W1 + b1)          (NT=4: 64-col groups)
  int waves1 = tiles_m * (HID / 64);
  k_gemm_wmma<4, false, true><<<(waves1 + 3) / 4, 128, 0, stream>>>(x, W1, b1, hA,
                                                                    Nn, IN, HID);
  // conv2 linear: hB = hA @ W2 + b2
  k_gemm_wmma<4, false, false><<<(waves1 + 3) / 4, 128, 0, stream>>>(hA, W2, b2, hB,
                                                                     Nn, HID, HID);

  // 30 propagation hops, ping-pong (L2-resident float4 gather, no atomics)
  int Fq = HID / 4;                        // 32
  const float* hin = hB; float* hout = hA;
  for (int h = 0; h < HOPS; ++h) {
    k_hop4<<<Nn, Fq, 0, stream>>>((const float4*)hin, (float4*)hout,
                                  row_ptr, csr_src, csr_nrm, Fq);
    const float* t = hout; hout = (float*)hin; hin = t;
  }

  // cls: h3 = relu(hin) @ Wc + bc   (relu fused into A-load; NT=1 since OUT=16)
  int waves3 = tiles_m * (OUT / 16);
  k_gemm_wmma<1, true, false><<<(waves3 + 3) / 4, 128, 0, stream>>>(hin, Wc, bc, h3,
                                                                    Nn, HID, OUT);

  // final single hop into d_out
  int Fq2 = OUT / 4;                       // 4
  int per = 32 / Fq2;                      // 8
  k_hop_small4<<<(Nn + per - 1) / per, 32, 0, stream>>>((const float4*)h3,
                                                        (float4*)d_out, row_ptr,
                                                        csr_src, csr_nrm, Nn, Fq2);
}
